// MoEHurricaneLSTM_70746701300282
// MI455X (gfx1250) — matrix operand order repaired and checked by reading, not compiled, not run on hardware
//
#include <hip/hip_runtime.h>
#include <math.h>

// ---------------------------------------------------------------------------
// Types for CDNA5 WMMA (wave32)
// ---------------------------------------------------------------------------
typedef __attribute__((ext_vector_type(16))) _Float16 v16h;
typedef __attribute__((ext_vector_type(8)))  _Float16 v8h;
typedef __attribute__((ext_vector_type(8)))  float    v8f;

#define Bn    2048
#define Hn    512
#define FourH 2048
#define Sn    64
#define En    8
#define HHn   256
#define HOR   8
#define NBIG  6400          // 256 (gate) + 3*2048 (expert banks)
#define OFF_G 0
#define OFF_T 256
#define OFF_I 2304
#define OFF_W 4352

// ---------------------------------------------------------------------------
// Fused dual-operand WMMA GEMM:
//   C[2048, N] (f32) = A1[2048,K1]*W1[N,K1]^T + A2[2048,K2]*W2[N,K2]^T
//                      [+bias] [relu]
// Block tile 64(M) x 128(N); 8 waves as 2(M) x 4(N); each wave owns a 32x32
// sub-tile -> 4 v_wmma_f32_16x16x32_f16 per K-step (2 A-frags x 2 B-frags).
// A2 == nullptr skips the second segment. flags: bit0 = relu.
// ---------------------------------------------------------------------------
__global__ __launch_bounds__(256)
void wmma_gemm2_k(const _Float16* __restrict__ A1, int lda1,
                  const _Float16* __restrict__ W1, int ldw1, int K1,
                  const _Float16* __restrict__ A2, int lda2,
                  const _Float16* __restrict__ W2, int ldw2, int K2,
                  float* __restrict__ C, int ldc,
                  const float* __restrict__ bias, int flags)
{
    __shared__ alignas(32) _Float16 As[64 * 32];
    __shared__ alignas(32) _Float16 Bs[128 * 32];

    const int tid  = threadIdx.x;
    const int wave = tid >> 5;
    const int lane = tid & 31;
    const int wmq  = wave & 1;      // 2 M groups of 32
    const int wnq  = wave >> 1;     // 4 N groups of 32
    const int half = lane >> 4;
    const int l16  = lane & 15;
    const int bm   = blockIdx.x * 64;
    const int bn   = blockIdx.y * 128;

    // cooperative tile load indices
    const int ar = tid >> 2;            // 0..63
    const int ac = (tid & 3) << 3;      // 0,8,16,24
    const int br = tid >> 1;            // 0..127
    const int bc = (tid & 1) << 4;      // 0,16

    v8f a00 = {}, a01 = {}, a10 = {}, a11 = {};

    #pragma unroll 1
    for (int s = 0; s < 2; ++s) {
        const _Float16* A = (s == 0) ? A1 : A2;
        if (!A) continue;
        const _Float16* W = (s == 0) ? W1 : W2;
        const int lda = (s == 0) ? lda1 : lda2;
        const int ldw = (s == 0) ? ldw1 : ldw2;
        const int K   = (s == 0) ? K1 : K2;

        const _Float16* Ag = A + (size_t)(bm + ar) * lda + ac;
        const _Float16* Wg = W + (size_t)(bn + br) * ldw + bc;

        for (int k0 = 0; k0 < K; k0 += 32) {
            __syncthreads();
            *reinterpret_cast<uint4*>(&As[ar * 32 + ac]) =
                *reinterpret_cast<const uint4*>(Ag + k0);
            *reinterpret_cast<uint4*>(&Bs[br * 32 + bc]) =
                *reinterpret_cast<const uint4*>(Wg + k0);
            *reinterpret_cast<uint4*>(&Bs[br * 32 + bc + 8]) =
                *reinterpret_cast<const uint4*>(Wg + k0 + 8);
            __syncthreads();

            // A fragments (contiguous k = 16*half + s, same order as B; M by lane)
            const int am0 = wmq * 32 + l16;
            v8h lo0 = *reinterpret_cast<const v8h*>(&As[ am0       * 32 +     half * 16]);
            v8h hi0 = *reinterpret_cast<const v8h*>(&As[ am0       * 32 + 8 + half * 16]);
            v8h lo1 = *reinterpret_cast<const v8h*>(&As[(am0 + 16) * 32 +     half * 16]);
            v8h hi1 = *reinterpret_cast<const v8h*>(&As[(am0 + 16) * 32 + 8 + half * 16]);
            v16h fa0 = __builtin_shufflevector(lo0, hi0, 0,1,2,3,4,5,6,7,8,9,10,11,12,13,14,15);
            v16h fa1 = __builtin_shufflevector(lo1, hi1, 0,1,2,3,4,5,6,7,8,9,10,11,12,13,14,15);

            // B fragments (N by lane 0-15, K split across lane halves)
            const int nb = wnq * 32 + l16;
            v16h fb0 = *reinterpret_cast<const v16h*>(&Bs[ nb       * 32 + half * 16]);
            v16h fb1 = *reinterpret_cast<const v16h*>(&Bs[(nb + 16) * 32 + half * 16]);

            a00 = __builtin_amdgcn_wmma_f32_16x16x32_f16(false, fa0, false, fb0, (short)0, a00, false, false);
            a01 = __builtin_amdgcn_wmma_f32_16x16x32_f16(false, fa0, false, fb1, (short)0, a01, false, false);
            a10 = __builtin_amdgcn_wmma_f32_16x16x32_f16(false, fa1, false, fb0, (short)0, a10, false, false);
            a11 = __builtin_amdgcn_wmma_f32_16x16x32_f16(false, fa1, false, fb1, (short)0, a11, false, false);
        }
    }

    // Epilogue (ISA C/D layout: VGPR i -> M = i, +8 for upper lane half)
    const int gm = bm + wmq * 32 + half * 8;
    const int gn = bn + wnq * 32 + l16;
    const float bv0 = bias ? bias[gn]      : 0.f;
    const float bv1 = bias ? bias[gn + 16] : 0.f;
    const bool relu = (flags & 1) != 0;
    #pragma unroll
    for (int i = 0; i < 8; ++i) {
        float v00 = a00[i] + bv0;
        float v01 = a01[i] + bv1;
        float v10 = a10[i] + bv0;
        float v11 = a11[i] + bv1;
        if (relu) {
            v00 = fmaxf(v00, 0.f); v01 = fmaxf(v01, 0.f);
            v10 = fmaxf(v10, 0.f); v11 = fmaxf(v11, 0.f);
        }
        float* r0 = C + (size_t)(gm + i)      * ldc;
        float* r1 = C + (size_t)(gm + 16 + i) * ldc;
        r0[gn] = v00; r0[gn + 16] = v01;
        r1[gn] = v10; r1[gn + 16] = v11;
    }
}

// ---------------------------------------------------------------------------
// Pointwise kernels
// ---------------------------------------------------------------------------
__device__ __forceinline__ float sigf(float x) { return 1.f / (1.f + __expf(-x)); }

__global__ void lstm_cell_k(const float* __restrict__ z,
                            float* __restrict__ c,
                            _Float16* __restrict__ h)
{
    int idx = blockIdx.x * 256 + threadIdx.x;           // B*H threads
    if (idx >= Bn * Hn) return;
    int b = idx >> 9;
    int j = idx & (Hn - 1);
    const float* zb = z + (size_t)b * FourH;
    float gi = sigf (zb[j]);
    float gf = sigf (zb[Hn + j]);
    float gg = tanhf(zb[2 * Hn + j]);
    float go = sigf (zb[3 * Hn + j]);
    float c2 = gf * c[idx] + gi * gg;
    float h2 = go * tanhf(c2);
    c[idx] = c2;
    h[idx] = (_Float16)h2;
}

__global__ void cvt_pad_k(const float* __restrict__ src, _Float16* __restrict__ dst,
                          int rows, int scols, int dcols)
{
    int i = blockIdx.x * 256 + threadIdx.x;
    if (i >= rows * dcols) return;
    int r = i / dcols, cc = i % dcols;
    dst[i] = (cc < scols) ? (_Float16)src[(size_t)r * scols + cc] : (_Float16)0.f;
}

// x (B,S,16) f32  ->  Xall (S,B,32) f16 zero-padded
__global__ void cvt_x_k(const float* __restrict__ x, _Float16* __restrict__ dst)
{
    int i = blockIdx.x * 256 + threadIdx.x;             // S*B*32
    if (i >= Sn * Bn * 32) return;
    int f  = i & 31;
    int tb = i >> 5;
    int b  = tb & (Bn - 1);
    int t  = tb >> 11;
    dst[i] = (f < 16) ? (_Float16)x[((size_t)b * Sn + t) * 16 + f] : (_Float16)0.f;
}

__global__ void zero_f32_k(float* p, int n)
{ int i = blockIdx.x * 256 + threadIdx.x; if (i < n) p[i] = 0.f; }

__global__ void zero_f16_k(_Float16* p, int n)
{ int i = blockIdx.x * 256 + threadIdx.x; if (i < n) p[i] = (_Float16)0.f; }

__global__ void concat_bias_k(const float* __restrict__ gb1, const float* __restrict__ tb1,
                              const float* __restrict__ ib1, const float* __restrict__ wb1,
                              float* __restrict__ dst)
{
    int i = blockIdx.x * 256 + threadIdx.x;
    if (i >= NBIG) return;
    float v;
    if      (i < OFF_T) v = gb1[i - OFF_G];
    else if (i < OFF_I) v = tb1[i - OFF_T];
    else if (i < OFF_W) v = ib1[i - OFF_I];
    else                v = wb1[i - OFF_W];
    dst[i] = v;
}

// out[b,n] = dot(A[b,:K], W[n,:K]) + bias[n]      (gate logits: N=8, K=256)
__global__ void dense_rowdot_k(const float* __restrict__ A, int lda,
                               const float* __restrict__ W, int ldw,
                               const float* __restrict__ bias,
                               float* __restrict__ out, int N, int K, int total)
{
    int i = blockIdx.x * 256 + threadIdx.x;
    if (i >= total) return;
    int b = i / N, n = i % N;
    const float* a = A + (size_t)b * lda;
    const float* w = W + (size_t)n * ldw;
    float s = bias[n];
    for (int k = 0; k < K; ++k) s += a[k] * w[k];
    out[i] = s;
}

__global__ void softmax8_k(const float* __restrict__ logits, float* __restrict__ gates)
{
    int b = blockIdx.x * 256 + threadIdx.x;
    if (b >= Bn) return;
    const float* l = logits + b * En;
    float m = l[0];
    for (int e = 1; e < En; ++e) m = fmaxf(m, l[e]);
    float s = 0.f, ex[En];
    for (int e = 0; e < En; ++e) { ex[e] = __expf(l[e] - m); s += ex[e]; }
    float inv = 1.f / s;
    for (int e = 0; e < En; ++e) gates[b * En + e] = ex[e] * inv;
}

// eout[b,e,o] = dot(hdd[b, e*256 : (e+1)*256], W2[e,o,:]) + b2[e,o]
__global__ void expert_out_k(const float* __restrict__ hdd, int ldh,
                             const float* __restrict__ W2,
                             const float* __restrict__ b2,
                             float* __restrict__ eout, int O, int total)
{
    int i = blockIdx.x * 256 + threadIdx.x;
    if (i >= total) return;
    int b = i / (En * O);
    int r = i % (En * O);
    int e = r / O, o = r % O;
    const float* hv = hdd + (size_t)b * ldh + e * HHn;
    const float* wv = W2 + ((size_t)e * O + o) * HHn;
    float s = b2[e * O + o];
    for (int k = 0; k < HHn; ++k) s += hv[k] * wv[k];
    eout[i] = s;
}

// Mixture combine; writes outputs for step t, accumulates gates_avg,
// builds next decoder input (f16, padded to 32).
__global__ void mix_step_k(const float* __restrict__ gates,
                           const float* __restrict__ et,
                           const float* __restrict__ ei,
                           const float* __restrict__ ew,
                           float* __restrict__ out,
                           _Float16* __restrict__ decin, int t)
{
    int b = blockIdx.x * 256 + threadIdx.x;
    if (b >= Bn) return;
    const float* g = gates + b * En;
    float t0 = 0.f, t1 = 0.f, dw = 0.f, wd = 0.f;
    for (int e = 0; e < En; ++e) {
        t0 += g[e] * et[b * (En * 2) + e * 2 + 0];
        t1 += g[e] * et[b * (En * 2) + e * 2 + 1];
        dw += g[e] * ei[b * En + e];
        wd += g[e] * ew[b * En + e];
    }
    out[b * 16 + t]          = t0;   // dlat
    out[b * 16 + 8 + t]      = t1;   // dlon
    out[Bn * 16 + b * 8 + t] = dw;   // intensity
    out[Bn * 24 + b * 8 + t] = wd;   // wind_abs
    for (int e = 0; e < En; ++e)
        out[Bn * 32 + b * 8 + e] += g[e] * (1.f / HOR);  // gates_avg
    _Float16* d = decin + b * 32;
    d[0] = (_Float16)t0; d[1] = (_Float16)t1;
    d[2] = (_Float16)dw; d[3] = (_Float16)wd;            // cols 4..31 stay zero
}

// ---------------------------------------------------------------------------
// Host orchestration
// ---------------------------------------------------------------------------
static inline int cdiv(int a, int b) { return (a + b - 1) / b; }

extern "C" void kernel_launch(void* const* d_in, const int* in_sizes, int n_in,
                              void* d_out, int out_size, void* d_ws, size_t ws_size,
                              hipStream_t stream)
{
    (void)in_sizes; (void)n_in; (void)out_size; (void)ws_size;
    const float* x     = (const float*)d_in[0];
    const float* eWih0 = (const float*)d_in[1];
    const float* eWhh0 = (const float*)d_in[2];
    const float* eb0   = (const float*)d_in[3];
    const float* eWih1 = (const float*)d_in[4];
    const float* eWhh1 = (const float*)d_in[5];
    const float* eb1   = (const float*)d_in[6];
    const float* dWih  = (const float*)d_in[7];
    const float* dWhh  = (const float*)d_in[8];
    const float* db    = (const float*)d_in[9];
    const float* gW1   = (const float*)d_in[10];
    const float* gb1   = (const float*)d_in[11];
    const float* gW2   = (const float*)d_in[12];
    const float* gb2   = (const float*)d_in[13];
    const float* tW1   = (const float*)d_in[14];
    const float* tb1   = (const float*)d_in[15];
    const float* tW2   = (const float*)d_in[16];
    const float* tb2   = (const float*)d_in[17];
    const float* iW1   = (const float*)d_in[18];
    const float* ib1   = (const float*)d_in[19];
    const float* iW2   = (const float*)d_in[20];
    const float* ib2   = (const float*)d_in[21];
    const float* wW1   = (const float*)d_in[22];
    const float* wb1   = (const float*)d_in[23];
    const float* wW2   = (const float*)d_in[24];
    const float* wb2   = (const float*)d_in[25];
    float* out = (float*)d_out;

    // workspace bump allocator (256B aligned)
    char* wsb = (char*)d_ws;
    size_t off = 0;
    auto alloc = [&](size_t bytes) -> char* {
        char* p = wsb + off;
        off = (off + bytes + 255) & ~(size_t)255;
        return p;
    };
    _Float16* wWih0 = (_Float16*)alloc((size_t)FourH * 32 * 2);
    _Float16* wWhh0 = (_Float16*)alloc((size_t)FourH * Hn * 2);
    _Float16* wWih1 = (_Float16*)alloc((size_t)FourH * Hn * 2);
    _Float16* wWhh1 = (_Float16*)alloc((size_t)FourH * Hn * 2);
    _Float16* wDWih = (_Float16*)alloc((size_t)FourH * 32 * 2);
    _Float16* wDWhh = (_Float16*)alloc((size_t)FourH * Hn * 2);
    _Float16* wBig  = (_Float16*)alloc((size_t)NBIG * Hn * 2);   // [gW1|tW1|iW1|wW1]
    float*    bBig  = (float*)   alloc((size_t)NBIG * 4);
    _Float16* Xall  = (_Float16*)alloc((size_t)Sn * Bn * 32 * 2);
    float*    z     = (float*)   alloc((size_t)Bn * FourH * 4);
    float*    c0    = (float*)   alloc((size_t)Bn * Hn * 4);
    float*    c1    = (float*)   alloc((size_t)Bn * Hn * 4);
    _Float16* h0    = (_Float16*)alloc((size_t)Bn * Hn * 2);
    _Float16* h1    = (_Float16*)alloc((size_t)Bn * Hn * 2);
    _Float16* decin = (_Float16*)alloc((size_t)Bn * 32 * 2);
    float*    ghdd  = (float*)   alloc((size_t)Bn * NBIG * 4);   // [g1|hdd_t|hdd_i|hdd_w]
    float*    lgts  = (float*)   alloc((size_t)Bn * En * 4);
    float*    gts   = (float*)   alloc((size_t)Bn * En * 4);
    float*    et    = (float*)   alloc((size_t)Bn * En * 2 * 4);
    float*    ei    = (float*)   alloc((size_t)Bn * En * 4);
    float*    ew    = (float*)   alloc((size_t)Bn * En * 4);

    // --- one-time conversions (f32 -> f16, K padded to multiple of 32) ---
    auto cvt = [&](const float* s, _Float16* d, int rows, int sc, int dc) {
        cvt_pad_k<<<cdiv(rows * dc, 256), 256, 0, stream>>>(s, d, rows, sc, dc);
    };
    cvt(eWih0, wWih0, FourH, 16, 32);
    cvt(eWhh0, wWhh0, FourH, Hn, Hn);
    cvt(eWih1, wWih1, FourH, Hn, Hn);
    cvt(eWhh1, wWhh1, FourH, Hn, Hn);
    cvt(dWih,  wDWih, FourH, 4, 32);
    cvt(dWhh,  wDWhh, FourH, Hn, Hn);
    cvt(gW1, wBig + (size_t)OFF_G * Hn, HHn,      Hn, Hn);
    cvt(tW1, wBig + (size_t)OFF_T * Hn, En * HHn, Hn, Hn);
    cvt(iW1, wBig + (size_t)OFF_I * Hn, En * HHn, Hn, Hn);
    cvt(wW1, wBig + (size_t)OFF_W * Hn, En * HHn, Hn, Hn);
    concat_bias_k<<<cdiv(NBIG, 256), 256, 0, stream>>>(gb1, tb1, ib1, wb1, bBig);
    cvt_x_k<<<cdiv(Sn * Bn * 32, 256), 256, 0, stream>>>(x, Xall);

    // --- state init (every call; deterministic) ---
    zero_f16_k<<<cdiv(Bn * Hn, 256), 256, 0, stream>>>(h0, Bn * Hn);
    zero_f16_k<<<cdiv(Bn * Hn, 256), 256, 0, stream>>>(h1, Bn * Hn);
    zero_f16_k<<<cdiv(Bn * 32, 256), 256, 0, stream>>>(decin, Bn * 32);
    zero_f32_k<<<cdiv(Bn * Hn, 256), 256, 0, stream>>>(c0, Bn * Hn);
    zero_f32_k<<<cdiv(Bn * Hn, 256), 256, 0, stream>>>(c1, Bn * Hn);
    zero_f32_k<<<cdiv(Bn * 40, 256), 256, 0, stream>>>(out, Bn * 40);

    auto gemm2 = [&](const _Float16* A1, int lda1, const _Float16* W1, int ldw1, int K1,
                     const _Float16* A2, int lda2, const _Float16* W2, int ldw2, int K2,
                     float* C, int N, const float* bias, int flags) {
        dim3 grid(Bn / 64, N / 128);
        wmma_gemm2_k<<<grid, 256, 0, stream>>>(A1, lda1, W1, ldw1, K1,
                                               A2, lda2, W2, ldw2, K2,
                                               C, N, bias, flags);
    };
    const int cellB = cdiv(Bn * Hn, 256);

    // --- encoder: two stacked LSTMs, pipelined per time step ---
    for (int t = 0; t < Sn; ++t) {
        // z0 = h0@Whh0^T + x_t@Wih0^T + b0   (single fused GEMM)
        gemm2(h0, Hn, wWhh0, Hn, Hn,
              Xall + (size_t)t * Bn * 32, 32, wWih0, 32, 32,
              z, FourH, eb0, 0);
        lstm_cell_k<<<cellB, 256, 0, stream>>>(z, c0, h0);
        // z1 = h0@Wih1^T + h1@Whh1^T + b1    (single fused GEMM)
        gemm2(h0, Hn, wWih1, Hn, Hn,
              h1, Hn, wWhh1, Hn, Hn,
              z, FourH, eb1, 0);
        lstm_cell_k<<<cellB, 256, 0, stream>>>(z, c1, h1);
    }

    // --- decoder: MoE autoregressive loop (h1/c1 carry over as h_n/c_n) ---
    for (int t = 0; t < HOR; ++t) {
        // z = h@dWhh^T + dec_in@dWih^T + db
        gemm2(h1, Hn, wDWhh, Hn, Hn,
              decin, 32, wDWih, 32, 32,
              z, FourH, db, 0);
        lstm_cell_k<<<cellB, 256, 0, stream>>>(z, c1, h1);     // ctx = h1

        // one GEMM: [gate hidden | track hdd | intensity hdd | wind hdd], relu
        gemm2(h1, Hn, wBig, Hn, Hn,
              nullptr, 0, nullptr, 0, 0,
              ghdd, NBIG, bBig, 1);

        dense_rowdot_k<<<cdiv(Bn * En, 256), 256, 0, stream>>>(
            ghdd + OFF_G, NBIG, gW2, HHn, gb2, lgts, En, HHn, Bn * En);
        softmax8_k<<<cdiv(Bn, 256), 256, 0, stream>>>(lgts, gts);

        expert_out_k<<<cdiv(Bn * En * 2, 256), 256, 0, stream>>>(
            ghdd + OFF_T, NBIG, tW2, tb2, et, 2, Bn * En * 2);
        expert_out_k<<<cdiv(Bn * En, 256), 256, 0, stream>>>(
            ghdd + OFF_I, NBIG, iW2, ib2, ei, 1, Bn * En);
        expert_out_k<<<cdiv(Bn * En, 256), 256, 0, stream>>>(
            ghdd + OFF_W, NBIG, wW2, wb2, ew, 1, Bn * En);

        mix_step_k<<<cdiv(Bn, 256), 256, 0, stream>>>(gts, et, ei, ew, out, decin, t);
    }
}
